// LinearAttention_6717328850976
// MI455X (gfx1250) — matrix operand order repaired and checked by reading, not compiled, not run on hardware
//
#include <hip/hip_runtime.h>
#include <hip/hip_bf16.h>

// ---------------------------------------------------------------------------
// LinearAttention on MI455X (gfx1250), wave32 + WMMA bf16.
//
// Pipeline (after algebraic fusion of the random-feature projection):
//   WqPt = Pt @ Wq_h^T (per head, emitted pre-transposed)   small WMMA GEMMs
//   Qk = relu(Xq @ WqP + bqP) + eps        [16384 x 2048]   bf16 WMMA GEMM
//   Kk = relu(Xk @ WkP + bkP) + eps        [16384 x 2048]   bf16 WMMA GEMM
//   V  = Xv @ Wv + bv                      [16384 x 2048]   bf16 WMMA GEMM
//   Ksum = sum_s Kk ; Srow = Ksum . V ; Z = rowsum(Qk)      small reductions
//   O  = Qk * Srow / (Z+eps)                                elementwise
//   out = O @ Wo + bo                      [16384 x 2048]   bf16 WMMA GEMM (f32)
//
// All GEMM B-operands are pre-transposed to [N x K] so both LDS tiles load
// as coalesced b128 pairs (no scalar ds_store transpose in the hot loop).
// ---------------------------------------------------------------------------

#define B_   64
#define S_   256
#define D_   2048
#define H_   8
#define DK_  256
#define F_   256
#define M_   (B_ * S_)          // 16384
#define EPS_ 1e-8f

typedef __attribute__((ext_vector_type(16))) __bf16 v16bf;
typedef __attribute__((ext_vector_type(8)))  __bf16 v8bf;
typedef __attribute__((ext_vector_type(8)))  float  v8f;

// ---------------------------------------------------------------------------
// Tiled bf16 WMMA GEMM:  C[M,N] = A[M,K] @ Bt[N,K]^T (+bias) (+epilogue)
//   EPI 0: f32 out, +bias
//   EPI 1: bf16 out, +bias, relu, +eps
//   EPI 2: bf16 out, +bias
//   EPI 3: bf16 out, no bias          (per-head weight folding)
// 256 threads (8 waves). BM=128 BN=128 BK=32, double-buffered LDS.
// Wave tile 32x64 (2x4 WMMA accumulators).
// Assumes M%128==0, N%128==0, K%32==0 (true for all call sites).
// ---------------------------------------------------------------------------
template <int EPI>
__global__ __launch_bounds__(256)
void gemm_bf16_wmma(const __bf16* __restrict__ A, const __bf16* __restrict__ Bt,
                    const float* __restrict__ bias, void* __restrict__ Cv,
                    int M, int N, int K, int lda, int ldbt, int ldc,
                    long batchOffA, long batchOffB, long batchOffC)
{
    __shared__ __bf16 sA[2][128 * 32];   // [row][k]  K-contiguous
    __shared__ __bf16 sB[2][128 * 32];   // [col][k]  K-contiguous

    const int t     = threadIdx.x;
    const int lane  = t & 31;
    const int wave  = t >> 5;            // 0..7
    const int waveM = wave & 3;          // 4 waves along M (32 rows each)
    const int waveN = wave >> 2;         // 2 waves along N (64 cols each)

    const int bm0 = blockIdx.y * 128;
    const int bn0 = blockIdx.x * 128;

    const __bf16* Ab = A  + (long)blockIdx.z * batchOffA;
    const __bf16* Bb = Bt + (long)blockIdx.z * batchOffB;

    v8f acc[2][4];
#pragma unroll
    for (int i = 0; i < 2; ++i)
#pragma unroll
        for (int j = 0; j < 4; ++j)
            acc[i][j] = (v8f){0.f, 0.f, 0.f, 0.f, 0.f, 0.f, 0.f, 0.f};

    // cooperative tile loads: thread -> one 128-row, 16-element K slab half
    const int row = t >> 1;              // 0..127
    const int kh  = (t & 1) * 16;        // 0 or 16
    const __bf16* aptr = Ab + (long)(bm0 + row) * lda  + kh;
    const __bf16* bptr = Bb + (long)(bn0 + row) * ldbt + kh;

    v8bf ra0, ra1, rb0, rb1;             // staging registers (global -> LDS)
    auto gload = [&](int k0) {
        ra0 = *(const v8bf*)(aptr + k0);
        ra1 = *(const v8bf*)(aptr + k0 + 8);
        rb0 = *(const v8bf*)(bptr + k0);
        rb1 = *(const v8bf*)(bptr + k0 + 8);
    };
    auto sstore = [&](int buf) {
        *(v8bf*)&sA[buf][row * 32 + kh]     = ra0;
        *(v8bf*)&sA[buf][row * 32 + kh + 8] = ra1;
        *(v8bf*)&sB[buf][row * 32 + kh]     = rb0;
        *(v8bf*)&sB[buf][row * 32 + kh + 8] = rb1;
    };

    const int nk = K >> 5;
    gload(0);
    sstore(0);
    if (nk > 1) gload(32);
    __syncthreads();

    const int rlo = lane & 15;
    const int sel = lane >> 4;

    for (int kt = 0; kt < nk; ++kt) {
        const int cur = kt & 1;
        if (kt + 1 < nk) {
            sstore(cur ^ 1);                       // tile kt+1 (regs already loaded)
            if (kt + 2 < nk) gload((kt + 2) << 5); // start tile kt+2
        }

        // ---- fragments: contiguous K per lane half (not ISA table order), same for A and B ----
        // A 16x32: lanes 0-15 row=lane K=0..15;         lanes 16-31 K=16..31
        // B 32x16: lanes 0-15 col=lane K=0..15;         lanes 16-31 K=16..31
        v16bf afrag[2], bfrag[4];
#pragma unroll
        for (int i = 0; i < 2; ++i) {
            int r  = waveM * 32 + i * 16 + rlo;
            int kb = sel * 16;
            v8bf lo = *(const v8bf*)&sA[cur][r * 32 + kb];
            v8bf hi = *(const v8bf*)&sA[cur][r * 32 + kb + 8];
#pragma unroll
            for (int e = 0; e < 8; ++e) { afrag[i][e] = lo[e]; afrag[i][e + 8] = hi[e]; }
        }
#pragma unroll
        for (int j = 0; j < 4; ++j) {
            int c  = waveN * 64 + j * 16 + rlo;
            int kb = sel * 16;
            v8bf lo = *(const v8bf*)&sB[cur][c * 32 + kb];
            v8bf hi = *(const v8bf*)&sB[cur][c * 32 + kb + 8];
#pragma unroll
            for (int e = 0; e < 8; ++e) { bfrag[j][e] = lo[e]; bfrag[j][e + 8] = hi[e]; }
        }

#pragma unroll
        for (int i = 0; i < 2; ++i)
#pragma unroll
            for (int j = 0; j < 4; ++j)
                acc[i][j] = __builtin_amdgcn_wmma_f32_16x16x32_bf16(
                    false, afrag[i], false, bfrag[j], (short)0, acc[i][j], false, false);
        __syncthreads();
    }

    // ---- epilogue: C 16x16 layout = VGPR r: lanes0-15 M=r, lanes16-31 M=r+8 ----
    const int chi = sel * 8;
    const long cbase = (long)blockIdx.z * batchOffC;
#pragma unroll
    for (int i = 0; i < 2; ++i) {
#pragma unroll
        for (int j = 0; j < 4; ++j) {
            int gm = bm0 + waveM * 32 + i * 16 + chi;
            int gn = bn0 + waveN * 64 + j * 16 + rlo;
            float bval = (EPI != 3) ? bias[gn] : 0.f;
#pragma unroll
            for (int r = 0; r < 8; ++r) {
                float c = acc[i][j][r] + bval;
                if (EPI == 1) { c = c > 0.f ? c : 0.f; c += EPS_; }
                long idx = cbase + (long)(gm + r) * ldc + gn;
                if (EPI == 0) ((float*)Cv)[idx]  = c;
                else          ((__bf16*)Cv)[idx] = (__bf16)c;
            }
        }
    }
}

// ---------------------------------------------------------------------------
// Small helper kernels
// ---------------------------------------------------------------------------
__global__ void f32_to_bf16_kernel(const float* __restrict__ s, __bf16* __restrict__ d, long n)
{
    long i = ((long)blockIdx.x * blockDim.x + threadIdx.x) * 4;
    if (i + 3 < n) {
        float4 v = *(const float4*)(s + i);
        d[i]     = (__bf16)v.x;
        d[i + 1] = (__bf16)v.y;
        d[i + 2] = (__bf16)v.z;
        d[i + 3] = (__bf16)v.w;
    }
}

// Wt[j, i] (bf16) = W[i, j] (f32).  R rows, C cols of W; both % 32 == 0.
__global__ __launch_bounds__(256)
void conv_transpose_kernel(const float* __restrict__ W, __bf16* __restrict__ Wt, int R, int C)
{
    __shared__ float tile[32][33];
    const int bi = blockIdx.y * 32, bj = blockIdx.x * 32;
    const int tx = threadIdx.x & 31, ty = threadIdx.x >> 5;   // 32 x 8
#pragma unroll
    for (int r = ty; r < 32; r += 8)
        tile[r][tx] = W[(long)(bi + r) * C + bj + tx];
    __syncthreads();
#pragma unroll
    for (int r = ty; r < 32; r += 8)
        Wt[(long)(bj + r) * R + bi + tx] = (__bf16)tile[tx][r];
}

// bdst[h*F+f] = sum_d bsrc[h*DK+d] * P[d*F+f]
__global__ void bias_proj_kernel(const float* __restrict__ bsrc, const float* __restrict__ P,
                                 float* __restrict__ bdst)
{
    int n = blockIdx.x * blockDim.x + threadIdx.x;       // 0..2047
    int h = n >> 8, f = n & 255;
    float acc = 0.f;
    for (int d = 0; d < DK_; ++d) acc += bsrc[h * DK_ + d] * P[d * F_ + f];
    bdst[n] = acc;
}

// Ksum[b, n] = sum_s Kk[(b*S+s)*D + n]
__global__ void ksum_kernel(const __bf16* __restrict__ Kk, float* __restrict__ Ksum)
{
    int idx = blockIdx.x * blockDim.x + threadIdx.x;     // 0 .. B*D-1
    int b = idx >> 11, n = idx & (D_ - 1);
    const __bf16* p = Kk + (long)b * S_ * D_ + n;
    float a = 0.f;
    for (int s = 0; s < S_; ++s) a += (float)p[(long)s * D_];
    Ksum[idx] = a;
}

// Srow[b, h*F+v] = sum_d Ksum[b, h*DK+d] * V[(b*S+v)*D + h*DK+d]
__global__ void srow_kernel(const float* __restrict__ Ksum, const __bf16* __restrict__ V,
                            float* __restrict__ Srow)
{
    int idx = blockIdx.x * blockDim.x + threadIdx.x;     // 0 .. B*H*S-1
    int b = idx >> 11;
    int rem = idx & 2047;
    int h = rem >> 8, v = rem & 255;
    const float*  ks = Ksum + b * D_ + h * DK_;
    const __bf16* vp = V + (long)(b * S_ + v) * D_ + h * DK_;
    float a = 0.f;
    for (int d = 0; d < DK_; ++d) a += ks[d] * (float)vp[d];
    Srow[b * D_ + h * F_ + v] = a;
}

// Z[m*H+h] = sum_f Qk[m*D + h*F + f]
__global__ void zrow_kernel(const __bf16* __restrict__ Qk, float* __restrict__ Z)
{
    int idx = blockIdx.x * blockDim.x + threadIdx.x;     // 0 .. M*H-1
    int m = idx >> 3, h = idx & 7;
    const __bf16* p = Qk + (long)m * D_ + h * F_;
    float a = 0.f;
    for (int f = 0; f < F_; ++f) a += (float)p[f];
    Z[idx] = a;
}

// O = Qk * Srow / (Z + eps)   (elementwise, bf16 out)
__global__ void scale_kernel(const __bf16* __restrict__ Qk, const float* __restrict__ Srow,
                             const float* __restrict__ Z, __bf16* __restrict__ O)
{
    long idx = (long)blockIdx.x * blockDim.x + threadIdx.x;   // 0 .. M*D-1
    long m = idx >> 11;
    int  n = (int)(idx & (D_ - 1));
    int  b = (int)(m >> 8);
    int  h = n >> 8;
    float q = (float)Qk[idx];
    O[idx] = (__bf16)(q * Srow[b * D_ + n] / (Z[m * H_ + h] + EPS_));
}

// ---------------------------------------------------------------------------
// Host-side orchestration
// ---------------------------------------------------------------------------
extern "C" void kernel_launch(void* const* d_in, const int* in_sizes, int n_in,
                              void* d_out, int out_size, void* d_ws, size_t ws_size,
                              hipStream_t stream)
{
    const float* q   = (const float*)d_in[0];
    const float* k   = (const float*)d_in[1];
    const float* v   = (const float*)d_in[2];
    const float* Wq  = (const float*)d_in[3];
    const float* bq  = (const float*)d_in[4];
    const float* Wk  = (const float*)d_in[5];
    const float* bk  = (const float*)d_in[6];
    const float* Wv  = (const float*)d_in[7];
    const float* bv  = (const float*)d_in[8];
    const float* Wo  = (const float*)d_in[9];
    const float* bo  = (const float*)d_in[10];
    const float* P   = (const float*)d_in[11];
    float* out       = (float*)d_out;

    // ---- workspace layout (256B aligned) ----
    char* ws = (char*)d_ws;
    size_t off = 0;
    auto alloc = [&](size_t bytes) -> size_t {
        size_t r = off;
        off += (bytes + 255) & ~(size_t)255;
        return r;
    };
    const size_t actB = (size_t)M_ * D_ * 2;       // bf16 activation matrix
    const size_t wB   = (size_t)D_ * D_ * 2;       // bf16 weight matrix

    size_t oXQ   = alloc(actB);                    // query bf16 (reused for O later)
    size_t oXK   = alloc(actB);
    size_t oXV   = alloc(actB);
    size_t oWQB  = alloc(wB);                      // Wq bf16 (plain, is Bt of fusion)
    size_t oWKB  = alloc(wB);
    size_t oWVT  = alloc(wB);                      // Wv^T bf16
    size_t oWOT  = alloc(wB);                      // Wo^T bf16
    size_t oPT   = alloc((size_t)DK_ * F_ * 2);    // P^T bf16
    size_t oWQPT = alloc(wB);                      // (Wq_h @ P)^T  per head
    size_t oWKPT = alloc(wB);
    size_t oBQP  = alloc((size_t)D_ * 4);
    size_t oBKP  = alloc((size_t)D_ * 4);
    size_t oQK   = alloc(actB);
    size_t oKK   = alloc(actB);
    size_t oVV   = alloc(actB);
    size_t oKSUM = alloc((size_t)B_ * D_ * 4);
    size_t oSROW = alloc((size_t)B_ * D_ * 4);
    size_t oZ    = alloc((size_t)M_ * H_ * 4);
    (void)ws_size; (void)in_sizes; (void)n_in; (void)out_size;

    __bf16* XQ   = (__bf16*)(ws + oXQ);
    __bf16* XK   = (__bf16*)(ws + oXK);
    __bf16* XV   = (__bf16*)(ws + oXV);
    __bf16* WQB  = (__bf16*)(ws + oWQB);
    __bf16* WKB  = (__bf16*)(ws + oWKB);
    __bf16* WVT  = (__bf16*)(ws + oWVT);
    __bf16* WOT  = (__bf16*)(ws + oWOT);
    __bf16* PT   = (__bf16*)(ws + oPT);
    __bf16* WQPT = (__bf16*)(ws + oWQPT);
    __bf16* WKPT = (__bf16*)(ws + oWKPT);
    float*  BQP  = (float*)(ws + oBQP);
    float*  BKP  = (float*)(ws + oBKP);
    __bf16* QK   = (__bf16*)(ws + oQK);
    __bf16* KK   = (__bf16*)(ws + oKK);
    __bf16* VV   = (__bf16*)(ws + oVV);
    float*  KSUM = (float*)(ws + oKSUM);
    float*  SROW = (float*)(ws + oSROW);
    float*  Z    = (float*)(ws + oZ);
    __bf16* OO   = XQ;   // reuse: Xq dead after Qk GEMM, O computed afterwards

    const long nAct = (long)M_ * D_;               // 33,554,432
    const long nW   = (long)D_ * D_;               // 4,194,304

    // 1) fp32 -> bf16 conversions (plain) and convert+transpose for B operands
    f32_to_bf16_kernel<<<(unsigned)(nAct / 4 / 256), 256, 0, stream>>>(q,  XQ,  nAct);
    f32_to_bf16_kernel<<<(unsigned)(nAct / 4 / 256), 256, 0, stream>>>(k,  XK,  nAct);
    f32_to_bf16_kernel<<<(unsigned)(nAct / 4 / 256), 256, 0, stream>>>(v,  XV,  nAct);
    f32_to_bf16_kernel<<<(unsigned)(nW / 4 / 256),   256, 0, stream>>>(Wq, WQB, nW);
    f32_to_bf16_kernel<<<(unsigned)(nW / 4 / 256),   256, 0, stream>>>(Wk, WKB, nW);
    conv_transpose_kernel<<<dim3(D_ / 32, D_ / 32), 256, 0, stream>>>(Wv, WVT, D_, D_);
    conv_transpose_kernel<<<dim3(D_ / 32, D_ / 32), 256, 0, stream>>>(Wo, WOT, D_, D_);
    conv_transpose_kernel<<<dim3(F_ / 32, DK_ / 32), 256, 0, stream>>>(P, PT, DK_, F_);

    // 2) fold random-feature projection into weights, emitted TRANSPOSED:
    //    WqPt[h*F+f, m] = sum_d P[d,f] * Wq[m, h*DK+d]
    //    GEMM: M'=F, N'=D, K'=DK ; A=Pt (lda=DK) ; Bt=Wq (ldbt=D, +h*DK) ;
    //    C rows offset by h*F.
    {
        dim3 grid(D_ / 128, F_ / 128, H_);
        gemm_bf16_wmma<3><<<grid, 256, 0, stream>>>(PT, WQB, nullptr, WQPT,
            F_, D_, DK_, DK_, D_, D_, /*offA=*/0, /*offB=*/DK_, /*offC=*/(long)F_ * D_);
        gemm_bf16_wmma<3><<<grid, 256, 0, stream>>>(PT, WKB, nullptr, WKPT,
            F_, D_, DK_, DK_, D_, D_, 0, DK_, (long)F_ * D_);
    }
    bias_proj_kernel<<<D_ / 256, 256, 0, stream>>>(bq, P, BQP);
    bias_proj_kernel<<<D_ / 256, 256, 0, stream>>>(bk, P, BKP);

    // 3) the three fused projection GEMMs (Bt = pre-transposed weights)
    dim3 gridBig(D_ / 128, M_ / 128, 1);
    gemm_bf16_wmma<1><<<gridBig, 256, 0, stream>>>(XQ, WQPT, BQP, QK,
        M_, D_, D_, D_, D_, D_, 0, 0, 0);                       // Qk = relu(.)+eps
    gemm_bf16_wmma<1><<<gridBig, 256, 0, stream>>>(XK, WKPT, BKP, KK,
        M_, D_, D_, D_, D_, D_, 0, 0, 0);                       // Kk = relu(.)+eps
    gemm_bf16_wmma<2><<<gridBig, 256, 0, stream>>>(XV, WVT, bv, VV,
        M_, D_, D_, D_, D_, D_, 0, 0, 0);                       // V

    // 4) reductions + elementwise scale
    ksum_kernel<<<(B_ * D_) / 256, 256, 0, stream>>>(KK, KSUM);
    srow_kernel<<<(B_ * H_ * S_) / 256, 256, 0, stream>>>(KSUM, VV, SROW);
    zrow_kernel<<<(M_ * H_) / 256, 256, 0, stream>>>(QK, Z);
    scale_kernel<<<(unsigned)(nAct / 256), 256, 0, stream>>>(QK, SROW, Z, OO);

    // 5) output projection -> f32 d_out
    gemm_bf16_wmma<0><<<gridBig, 256, 0, stream>>>(OO, WOT, bo, out,
        M_, D_, D_, D_, D_, D_, 0, 0, 0);
}
